// liGRU_21311627723483
// MI455X (gfx1250) — compile-verified, matrix operand order checked by reading, not run on hardware
//
#include <hip/hip_runtime.h>
#include <hip/hip_bf16.h>

// ---------------------------------------------------------------------------
// liGRU (bidirectional) for MI455X / gfx1250.
//   T=2000, B=32 (=> 64 concat batch rows), F=H=512.
// Phase 1: BN(x) via deterministic 2-stage coalesced reduction -> bf16 acts
// Phase 2: WMMA bf16 GEMMs (4 N-tiles/wave, double-buffered fragments,
//          single-base addressing so offsets fold into load immediates)
// Phase 3: ONE persistent scan kernel (16 co-resident blocks) iterating all
//          2000 steps with a device-wide atomic barrier: both gates via
//          v_wmma_f32_16x16x32_bf16, h carried in registers across steps,
//          gate-BN in-register, next-step gate rows prefetched, writes
//          straight into the final (T,32,1024) output.
// ---------------------------------------------------------------------------

typedef __attribute__((ext_vector_type(16))) __bf16 v16bf;
typedef __attribute__((ext_vector_type(8)))  float  v8f;
typedef __attribute__((ext_vector_type(8)))  unsigned int v8u;

#define T_STEPS 2000
#define B_SZ    32
#define FH      512
#define NROW    (T_STEPS * B_SZ)   // 64000 forward rows
#define NB_STAT 500                // stats stage-1 blocks (64000/500=128 rows)
#define ROWS_PB 128
#define SCAN_BLOCKS 16             // must be co-resident (MI455X has >>16 WGPs)
#define U_OFF   (FH * FH)          // Uz offset inside combined U buffer (elems)

__device__ __forceinline__ unsigned short f2bf_u(float f) {
  unsigned u = __builtin_bit_cast(unsigned, f);
  u += 0x7fffu + ((u >> 16) & 1u);           // round-to-nearest-even
  return (unsigned short)(u >> 16);
}

// Load one lane's 16 bf16 fragment values. p0 already includes the lane-half
// base (+8*khalf); K order per lane-half is {0..7,16..23}/{8..15,24..31}.
__device__ __forceinline__ v16bf load_frag(const unsigned short* __restrict__ p0,
                                           int kbase) {
  const unsigned int* p = (const unsigned int*)(p0 + kbase);
  v8u v;
#pragma unroll
  for (int q = 0; q < 4; ++q) v[q] = p[q];         // K offsets +0..+7
#pragma unroll
  for (int q = 0; q < 4; ++q) v[4 + q] = p[8 + q]; // K offsets +16..+23
  return __builtin_bit_cast(v16bf, v);
}

// ---------------- BatchNorm: deterministic 2-stage reduction ----------------
__global__ void bn_partial(const float* __restrict__ X,
                           float* __restrict__ psum, float* __restrict__ psq) {
  int blk = blockIdx.x, t = threadIdx.x;          // 256 threads
  const float* base = X + (size_t)blk * ROWS_PB * FH;
  float s0 = 0.f, q0 = 0.f, s1 = 0.f, q1 = 0.f;
#pragma unroll 4
  for (int r = 0; r < ROWS_PB; ++r) {
    float a = base[r * FH + t];
    float b = base[r * FH + t + 256];
    s0 += a; q0 += a * a;
    s1 += b; q1 += b * b;
  }
  psum[(size_t)blk * FH + t]       = s0;
  psum[(size_t)blk * FH + t + 256] = s1;
  psq [(size_t)blk * FH + t]       = q0;
  psq [(size_t)blk * FH + t + 256] = q1;
}

// Emits fused scale/shift: y = x*sc + sh, sc = g*rsqrt(var+eps), sh = b - m*sc
__global__ void bn_finalize(const float* __restrict__ psum, const float* __restrict__ psq,
                            const float* __restrict__ g, const float* __restrict__ b,
                            float* __restrict__ sc, float* __restrict__ sh) {
  int f = blockIdx.x * blockDim.x + threadIdx.x;
  if (f >= FH) return;
  float s = 0.f, q = 0.f;
  for (int p = 0; p < NB_STAT; ++p) {
    s += psum[(size_t)p * FH + f];
    q += psq [(size_t)p * FH + f];
  }
  float inv = 1.0f / (float)NROW;
  float m   = s * inv;
  float var = q * inv - m * m;
  float r   = rsqrtf(var + 1e-5f);
  float scale = g[f] * r;
  sc[f] = scale;
  sh[f] = b[f] - m * scale;
}

__global__ void bn_apply_bf16(const float* __restrict__ X, unsigned short* __restrict__ Y,
                              const float* __restrict__ sc, const float* __restrict__ sh,
                              int n) {
  int i = blockIdx.x * blockDim.x + threadIdx.x;
  if (i < n) { int f = i & (FH - 1); Y[i] = f2bf_u(X[i] * sc[f] + sh[f]); }
}

__global__ void cvt_bf16(const float* __restrict__ X, unsigned short* __restrict__ Y, int n) {
  int i = blockIdx.x * blockDim.x + threadIdx.x;
  if (i < n) Y[i] = f2bf_u(X[i]);
}

__global__ void init_h(unsigned short* __restrict__ hbf0, unsigned short* __restrict__ hbf1,
                       int n, unsigned* __restrict__ ctr) {
  int i = blockIdx.x * blockDim.x + threadIdx.x;
  if (i < n) { hbf0[i] = 0; hbf1[i] = 0; }
  if (i == 0) *ctr = 0u;   // barrier counter for the persistent scan
}

// ---------------- Phase 2: WMMA GEMM, 1 M-tile x 4 N-tiles per wave ---------
// Double buffering + single-base addressing (B rows are j*16*FH elements
// apart -> folds into the 24-bit load immediate; no pointer spill).
__global__ void gemm_xw(const unsigned short* __restrict__ A,   // NROW x 512 bf16
                        const unsigned short* __restrict__ W,   // 512  x 512 bf16 (h-major)
                        float* __restrict__ C, int mtiles) {
  int wave = (int)((blockIdx.x * blockDim.x + threadIdx.x) >> 5);
  int lane = threadIdx.x & 31;
  int tm  = wave >> 3;          // 8 waves per M row (4 N tiles each)
  int tn0 = (wave & 7) * 4;
  if (tm >= mtiles) return;     // wave-uniform: EXEC stays all-ones for WMMA
  int r = lane & 15, khalf = lane >> 4;
  const unsigned short* arow  = A + (size_t)(tm * 16 + r) * FH + 8 * khalf;
  const unsigned short* bbase = W + (size_t)(tn0 * 16 + r) * FH + 8 * khalf;

  v8f acc[4] = {};
  v16bf a = load_frag(arow, 0);
  v16bf b[4];
#pragma unroll
  for (int j = 0; j < 4; ++j) b[j] = load_frag(bbase + j * 16 * FH, 0);

#pragma unroll 2
  for (int kb = 0; kb < FH; kb += 32) {
    v16bf an = {};
    v16bf bn[4] = {};
    int kn = kb + 32;
    if (kn < FH) {                       // uniform condition
      an = load_frag(arow, kn);
#pragma unroll
      for (int j = 0; j < 4; ++j) bn[j] = load_frag(bbase + j * 16 * FH, kn);
    }
#pragma unroll
    for (int j = 0; j < 4; ++j)
      acc[j] = __builtin_amdgcn_wmma_f32_16x16x32_bf16(false, a, false, b[j],
                                                       (short)0, acc[j], false, false);
    a = an;
#pragma unroll
    for (int j = 0; j < 4; ++j) b[j] = bn[j];
  }

  // C/D layout: VGPR rr -> row (tm*16 + rr + 8*khalf), col = tile*16 + lane&15
  float* cbase = C + (size_t)(tm * 16 + 8 * khalf) * FH + tn0 * 16 + (lane & 15);
#pragma unroll
  for (int j = 0; j < 4; ++j)
#pragma unroll
    for (int rr = 0; rr < 8; ++rr)
      cbase[rr * FH + j * 16] = acc[j][rr];
}

// ---------------- Phase 3: persistent scan kernel ---------------------------
// 16 blocks x 8 waves = 128 waves = 4 M-tiles x 32 N-tiles over the 64x512
// state. One kernel iterates all 2000 steps; blocks synchronize per step via
// release-fence -> monotonic atomic counter -> acquire-fence. Each wave
// carries its own 8 h-cells/lane in registers across all steps (the wave that
// writes a cell is the only one that reads it in fp32).
__global__ void ligru_scan(unsigned short* __restrict__ h0bf,
                           unsigned short* __restrict__ h1bf,
                           const float* __restrict__ WH, const float* __restrict__ WZ,
                           const unsigned short* __restrict__ U,   // [Uh | Uz] bf16
                           const float* __restrict__ scwh, const float* __restrict__ shwh,
                           const float* __restrict__ scwz, const float* __restrict__ shwz,
                           float* __restrict__ out, unsigned* __restrict__ ctr) {
  int wave = (int)((blockIdx.x * blockDim.x + threadIdx.x) >> 5); // 0..127
  int lane = threadIdx.x & 31;
  int tn = wave & 31, tm = wave >> 5;                             // tm 0..3
  int r = lane & 15, khalf = lane >> 4;
  const unsigned short* ubase = U + (size_t)(tn * 16 + r) * FH + 8 * khalf;
  const size_t arow_off = (size_t)(tm * 16 + r) * FH + 8 * khalf;
  int col = tn * 16 + (lane & 15);
  float sch = scwh[col], shh = shwh[col];      // gate BN folded into the scan
  float scz = scwz[col], shz = shwz[col];
  const bool fwd = (tm < 2);                   // rowm<32 <=> tm<2 (wave-uniform)
  const int rowbase = tm * 16 + 8 * khalf;     // epilogue rows rowbase..+7
  const int batbase = fwd ? rowbase : rowbase - B_SZ;

  v8f hreg = {};                               // h state, lives in registers
  for (int t = 0; t < T_STEPS; ++t) {
    const unsigned short* arow = ((t & 1) ? h1bf : h0bf) + arow_off;
    unsigned short* hnext_bf   = ((t & 1) ? h0bf : h1bf);

    v8f ah0 = {}, ah1 = {}, az0 = {}, az1 = {};
#pragma unroll 2
    for (int kb = 0; kb < FH; kb += 64) {
      v16bf a0  = load_frag(arow,  kb);
      v16bf a1  = load_frag(arow,  kb + 32);
      v16bf bh0 = load_frag(ubase, kb);
      v16bf bz0 = load_frag(ubase + U_OFF, kb);
      v16bf bh1 = load_frag(ubase, kb + 32);
      v16bf bz1 = load_frag(ubase + U_OFF, kb + 32);
      ah0 = __builtin_amdgcn_wmma_f32_16x16x32_bf16(false, a0, false, bh0, (short)0, ah0, false, false);
      az0 = __builtin_amdgcn_wmma_f32_16x16x32_bf16(false, a0, false, bz0, (short)0, az0, false, false);
      ah1 = __builtin_amdgcn_wmma_f32_16x16x32_bf16(false, a1, false, bh1, (short)0, ah1, false, false);
      az1 = __builtin_amdgcn_wmma_f32_16x16x32_bf16(false, a1, false, bz1, (short)0, az1, false, false);
    }

    // Time index of this wave's data (forward t, backward T-1-t) + next step's
    int td  = fwd ? t : (T_STEPS - 1) - t;
    int tn1 = (t + 1 < T_STEPS) ? t + 1 : t;
    int tdn = fwd ? tn1 : (T_STEPS - 1) - tn1;
    const float* wzrow = WZ + ((size_t)td * B_SZ + batbase) * FH + col;
    const float* whrow = WH + ((size_t)td * B_SZ + batbase) * FH + col;
    float* orow = out + ((size_t)td * B_SZ + batbase) * (2 * FH) + (fwd ? 0 : FH) + col;
    unsigned short* hrow = hnext_bf + (size_t)rowbase * FH + col;

#pragma unroll
    for (int rr = 0; rr < 8; ++rr) {
      float accZ = az0[rr] + az1[rr];
      float accH = ah0[rr] + ah1[rr];
      float wzv = wzrow[rr * FH] * scz + shz;   // BN(wz) in-register
      float whv = whrow[rr * FH] * sch + shh;   // BN(wh) in-register
      float z  = 1.0f / (1.0f + __expf(-(wzv + accZ)));
      float hc = whv + accH;
      hc = (hc > 0.f ? hc : 0.f) * 0.8f;        // relu * DROP_KEEP
      float hn = z * hreg[rr] + (1.0f - z) * hc;
      hreg[rr] = hn;                            // carried to next step
      hrow[rr * FH] = f2bf_u(hn);
      orow[rr * 2 * FH] = hn;
    }
    // Warm L2 for the next step's gate rows (stream-once HBM data).
    if ((lane & 15) == 0) {
      const float* wzn = WZ + ((size_t)tdn * B_SZ + batbase) * FH + col;
      const float* whn = WH + ((size_t)tdn * B_SZ + batbase) * FH + col;
#pragma unroll
      for (int rr = 0; rr < 8; ++rr) {
        __builtin_prefetch(&wzn[rr * FH], 0, 1);
        __builtin_prefetch(&whn[rr * FH], 0, 1);
      }
    }

    // ---- device-wide barrier (16 co-resident blocks) ----
    __threadfence();                 // release: my h stores visible device-wide
    __syncthreads();
    if (threadIdx.x == 0) {
      atomicAdd(ctr, 1u);
      unsigned target = (unsigned)gridDim.x * (unsigned)(t + 1);
      while (atomicAdd(ctr, 0u) < target) {
        __builtin_amdgcn_s_sleep(1);
      }
    }
    __syncthreads();
    __threadfence();                 // acquire: drop stale cached h lines
  }
}

// ---------------------------------------------------------------------------

extern "C" void kernel_launch(void* const* d_in, const int* in_sizes, int n_in,
                              void* d_out, int out_size, void* d_ws, size_t ws_size,
                              hipStream_t stream) {
  (void)in_sizes; (void)n_in; (void)out_size; (void)ws_size;
  const float* x    = (const float*)d_in[0];
  const float* Wh   = (const float*)d_in[1];
  const float* Wz   = (const float*)d_in[2];
  const float* Uh   = (const float*)d_in[3];
  const float* Uz   = (const float*)d_in[4];
  const float* g0   = (const float*)d_in[5];
  const float* b0   = (const float*)d_in[6];
  const float* g_wh = (const float*)d_in[7];
  const float* b_wh = (const float*)d_in[8];
  const float* g_wz = (const float*)d_in[9];
  const float* b_wz = (const float*)d_in[10];
  float* out = (float*)d_out;

  size_t off = 0;
  auto wsalloc = [&](size_t bytes) -> void* {
    void* p = (char*)d_ws + off;
    off += (bytes + 255) & ~(size_t)255;
    return p;
  };
  const size_t NE = (size_t)NROW * FH;                 // 32.77M elements
  unsigned short* xbf  = (unsigned short*)wsalloc(NE * 2);
  float*          WHf  = (float*)wsalloc(NE * 4);
  float*          WZf  = (float*)wsalloc(NE * 4);
  unsigned short* Whbf = (unsigned short*)wsalloc((size_t)FH * FH * 2);
  unsigned short* Wzbf = (unsigned short*)wsalloc((size_t)FH * FH * 2);
  unsigned short* Ubf  = (unsigned short*)wsalloc((size_t)2 * FH * FH * 2); // [Uh|Uz]
  unsigned short* h0bf = (unsigned short*)wsalloc((size_t)64 * FH * 2);
  unsigned short* h1bf = (unsigned short*)wsalloc((size_t)64 * FH * 2);
  float* psum = (float*)wsalloc((size_t)NB_STAT * FH * 4);
  float* psq  = (float*)wsalloc((size_t)NB_STAT * FH * 4);
  float* scx  = (float*)wsalloc(FH * 4);
  float* shx  = (float*)wsalloc(FH * 4);
  float* scwh = (float*)wsalloc(FH * 4);
  float* shwh = (float*)wsalloc(FH * 4);
  float* scwz = (float*)wsalloc(FH * 4);
  float* shwz = (float*)wsalloc(FH * 4);
  unsigned* ctr = (unsigned*)wsalloc(256);

  const int TPB = 256;
  const int nEl = (int)NE;
  const int wEl = FH * FH;

  // Phase 1: BN(x) -> bf16 (two-stage deterministic, coalesced stats)
  bn_partial<<<NB_STAT, TPB, 0, stream>>>(x, psum, psq);
  bn_finalize<<<2, TPB, 0, stream>>>(psum, psq, g0, b0, scx, shx);
  bn_apply_bf16<<<(nEl + TPB - 1) / TPB, TPB, 0, stream>>>(x, xbf, scx, shx, nEl);

  // Weight conversions to bf16 (Uh and Uz packed into one buffer)
  cvt_bf16<<<(wEl + TPB - 1) / TPB, TPB, 0, stream>>>(Wh, Whbf, wEl);
  cvt_bf16<<<(wEl + TPB - 1) / TPB, TPB, 0, stream>>>(Wz, Wzbf, wEl);
  cvt_bf16<<<(wEl + TPB - 1) / TPB, TPB, 0, stream>>>(Uh, Ubf, wEl);
  cvt_bf16<<<(wEl + TPB - 1) / TPB, TPB, 0, stream>>>(Uz, Ubf + U_OFF, wEl);

  // Phase 2: big WMMA GEMMs (forward rows only; backward half is a
  // permutation, handled by indexing in the scan) + gate-BN stats.
  const int mtiles = NROW / 16;                // 4000
  gemm_xw<<<mtiles, TPB, 0, stream>>>(xbf, Whbf, WHf, mtiles);
  gemm_xw<<<mtiles, TPB, 0, stream>>>(xbf, Wzbf, WZf, mtiles);
  bn_partial<<<NB_STAT, TPB, 0, stream>>>(WHf, psum, psq);
  bn_finalize<<<2, TPB, 0, stream>>>(psum, psq, g_wh, b_wh, scwh, shwh);
  bn_partial<<<NB_STAT, TPB, 0, stream>>>(WZf, psum, psq);
  bn_finalize<<<2, TPB, 0, stream>>>(psum, psq, g_wz, b_wz, scwz, shwz);

  // Phase 3: single persistent scan kernel (one graph node, no per-step
  // launch overhead); gate BN applied inside; h carried in registers.
  init_h<<<(64 * FH + TPB - 1) / TPB, TPB, 0, stream>>>(h0bf, h1bf, 64 * FH, ctr);
  ligru_scan<<<SCAN_BLOCKS, TPB, 0, stream>>>(h0bf, h1bf, WHf, WZf, Ubf,
                                              scwh, shwh, scwz, shwz, out, ctr);
}
